// DualArmActionHead_82231443849576
// MI455X (gfx1250) — compile-verified, matrix operand order checked by reading, not run on hardware
//
#include <hip/hip_runtime.h>
#include <hip/hip_bf16.h>

// ---------------------------------------------------------------------------
// DualArmActionHead on MI455X (gfx1250): 24-block scan, bf16 WMMA GEMM core.
// fp32 weights streamed from HBM (bandwidth-bound). fp32 -> bf16 conversion is
// done ONCE at LDS staging; fragments are then pure ds_load_b128 of packed
// bf16 pairs feeding v_wmma_f32_16x16x32_bf16 (fp32 accumulate).
// ---------------------------------------------------------------------------

typedef __attribute__((ext_vector_type(16))) __bf16 v16bf;
typedef __attribute__((ext_vector_type(8)))  float  v8f;

// problem constants
constexpr int D    = 512;
constexpr int NBLK = 24;
constexpr int BB   = 8;     // batch
constexpr int NAC  = 8;     // action tokens
constexpr int MQ   = BB * NAC;   // 64 rows
constexpr int MT   = BB * 64;    // 512 rows (h_t)
constexpr int KCTX = 88;         // 8 + 8 + 64 + 8

// workspace layout (floats), per arm
constexpr size_t OQ    = 0;
constexpr size_t OKS   = OQ    + (size_t)MQ * D;
constexpr size_t OVS   = OKS   + (size_t)MQ * D;
constexpr size_t OKT   = OVS   + (size_t)MQ * D;
constexpr size_t OVT   = OKT   + (size_t)MT * D;
constexpr size_t OKA   = OVT   + (size_t)MT * D;
constexpr size_t OVA   = OKA   + (size_t)MQ * D;
constexpr size_t OKI   = OVA   + (size_t)MQ * D;
constexpr size_t OVI   = OKI   + (size_t)MQ * D;
constexpr size_t OAOUT = OVI   + (size_t)MQ * D;
constexpr size_t OYBUF = OAOUT + (size_t)MQ * D;
constexpr size_t OLNY  = OYBUF + (size_t)MQ * D;
constexpr size_t ARM_STR = OLNY + (size_t)MQ * D;   // 851968 floats / arm
constexpr size_t XSZ   = (size_t)MQ * D;            // 32768 floats

// pack two fp32 into two bf16 (RNE) in one u32: lo = a, hi = b
__device__ inline unsigned int pack_bf2(float a, float b) {
#if __has_builtin(__builtin_amdgcn_cvt_pk_bf16_f32)
  auto r = __builtin_amdgcn_cvt_pk_bf16_f32(a, b);   // v_cvt_pk_bf16_f32
  return __builtin_bit_cast(unsigned int, r);
#else
  unsigned int ua = __float_as_uint(a);
  unsigned int ub = __float_as_uint(b);
  ua = (ua + 0x7FFFu + ((ua >> 16) & 1u)) >> 16;     // RNE to bf16
  ub = (ub + 0x7FFFu + ((ub >> 16) & 1u)) >> 16;
  return (ua & 0xFFFFu) | (ub << 16);
#endif
}

union Frag { v16bf v; unsigned int u[8]; uint4 q[2]; };

// LDS strides (in u32 units): 16 packed k-pairs per row + pad to 20
// (multiple of 4 -> uint4 loads stay 16B aligned; 20*n mod 64 is injective
//  over 16 consecutive n -> conflict-free fragment reads per 16-lane group)
constexpr int LBS = 20;

// ---------------------------------------------------------------------------
// GEMM core: block = 128 threads (4 wave32). Block computes a 16-row x 256-col
// output tile of  dst = A(16xK=512) @ W(512x512)[:, colBase:colBase+256] + bias
// with optional residual add and ReLU. Wave w owns cols [w*64, w*64+64).
// A/B converted fp32->bf16 at staging, stored as packed k-pairs so fragment
// builds are two ds_load_b128 each (matches CDNA5 16-bit WMMA layout 7.12.2:
// lanes 0-15 hold K 0-7 / 16-23 halves, lanes 16-31 hold K 8-15 / 24-31).
// ---------------------------------------------------------------------------
template <bool RELU, bool RESID>
__device__ void gemm_core(const float* const* sRow, const float* __restrict__ W,
                          const float* __restrict__ bias, const float* __restrict__ resid,
                          float* __restrict__ dst, int colBase)
{
  __shared__ __align__(16) unsigned int lAp[16 * LBS];
  __shared__ __align__(16) unsigned int lBp[256 * LBS];

  const int tid  = threadIdx.x;
  const int wave = tid >> 5;
  const int lane = tid & 31;
  const int g    = lane >> 4;    // K-half group (ISA: lanes 16-31 hold K+8)
  const int m    = lane & 15;    // A row / B column within tile

  v8f acc[4];
#pragma unroll
  for (int ti = 0; ti < 4; ++ti)
#pragma unroll
    for (int e = 0; e < 8; ++e) acc[ti][e] = 0.0f;

  for (int kk = 0; kk < 16; ++kk) {   // K = 512 in steps of 32
    __syncthreads();
    // ---- stage A tile (16 x 32 f32 -> packed bf16 pairs) ----
    {
      const int r  = tid >> 3;
      const int k4 = (tid & 7) * 4;
      const float4 va = *(const float4*)(sRow[r] + kk * 32 + k4);
      const int kp = k4 >> 1;
      lAp[r * LBS + kp]     = pack_bf2(va.x, va.y);
      lAp[r * LBS + kp + 1] = pack_bf2(va.z, va.w);
    }
    // ---- stage B slice (32 x 256 of W) -> lBp[n][k/2], packed along k ----
#pragma unroll
    for (int r = 0; r < 8; ++r) {
      const int k2 = r * 4 + (tid >> 6) * 2;          // even k: 0,2,...,30
      const int n  = (tid & 63) * 4;
      const float4 va = *(const float4*)(W + (size_t)(kk * 32 + k2    ) * D + colBase + n);
      const float4 vb = *(const float4*)(W + (size_t)(kk * 32 + k2 + 1) * D + colBase + n);
      const int kp = k2 >> 1;
      lBp[(n + 0) * LBS + kp] = pack_bf2(va.x, vb.x);
      lBp[(n + 1) * LBS + kp] = pack_bf2(va.y, vb.y);
      lBp[(n + 2) * LBS + kp] = pack_bf2(va.z, vb.z);
      lBp[(n + 3) * LBS + kp] = pack_bf2(va.w, vb.w);
    }
    if (kk + 1 < 16)   // pull next W slice toward L2 (global_prefetch_b8)
      __builtin_prefetch(W + (size_t)((kk + 1) * 32 + (tid >> 6)) * D + colBase + (tid & 63) * 4, 0, 1);
    __syncthreads();

    // ---- A fragment: two b128 LDS loads, zero VALU ----
    Frag a;
    a.q[0] = *(const uint4*)&lAp[m * LBS + g * 4];
    a.q[1] = *(const uint4*)&lAp[m * LBS + 8 + g * 4];

    // ---- 4 B fragments + WMMA ----
#pragma unroll
    for (int ti = 0; ti < 4; ++ti) {
      const int nb = (wave * 64 + ti * 16 + m) * LBS;
      Frag bfr;
      bfr.q[0] = *(const uint4*)&lBp[nb + g * 4];
      bfr.q[1] = *(const uint4*)&lBp[nb + 8 + g * 4];
      acc[ti] = __builtin_amdgcn_wmma_f32_16x16x32_bf16(
          false, a.v, false, bfr.v, (short)0, acc[ti], false, false);
    }
  }

  // ---- epilogue: C layout (7.12.2): lane m = N, vgpr e -> M = e + 8*g ----
#pragma unroll
  for (int ti = 0; ti < 4; ++ti) {
    const int n  = colBase + wave * 64 + ti * 16 + m;
    const float bv = bias[n];
#pragma unroll
    for (int e = 0; e < 8; ++e) {
      const int mm = e + g * 8;
      float val = acc[ti][e] + bv;
      if (RESID) val += resid[(size_t)mm * D + n];
      if (RELU)  val = fmaxf(val, 0.0f);
      dst[(size_t)mm * D + n] = val;
    }
  }
}

// ---------------------------------------------------------------------------
// 9 projections (q,ks,vs,kt,vt,ka,va,ki,vi) for both arms in one launch.
// grid.x = 92 row-tile slots, grid.y = column half (2x256), grid.z = arm.
// ---------------------------------------------------------------------------
__global__ __launch_bounds__(128)
void proj9_kernel(const float* __restrict__ ahs,
                  const float* __restrict__ WL, const float* __restrict__ WR,
                  const float* __restrict__ bL, const float* __restrict__ bR,
                  const float* __restrict__ xL, const float* __restrict__ xR,
                  float* __restrict__ ws, int t)
{
  const int arm = blockIdx.z;
  const int colBase = blockIdx.y * 256;
  const int pstart[10] = {0, 4, 8, 12, 44, 76, 80, 84, 88, 92};
  int p = 0, bx = blockIdx.x;
  while (p < 8 && bx >= pstart[p + 1]) ++p;
  const int rt = bx - pstart[p];

  const float* W    = (arm ? WR : WL) + (size_t)(t * 11 + p) * D * D;
  const float* bias = (arm ? bR : bL) + (size_t)(t * 11 + p) * D;
  const size_t dstoff[9] = {OQ, OKS, OVS, OKT, OVT, OKA, OVA, OKI, OVI};
  float* dst = ws + (size_t)arm * ARM_STR + dstoff[p] + (size_t)rt * 16 * D;

  const float* xq = arm ? xR : xL;
  const float* xo = arm ? xL : xR;

  __shared__ const float* sRow[16];
  if (threadIdx.x < 16) {
    const int row = rt * 16 + threadIdx.x;
    const float* src;
    if (p < 3) {
      src = xq + (size_t)row * D;                                 // x_self
    } else if (p < 5) {
      const int b = row >> 6, pos = row & 63;                     // h_t
      src = ahs + ((size_t)(b * 25 + (t + 1)) * 80 + pos) * D;
    } else if (p < 7) {
      const int b = row >> 3, pos = row & 7;                      // h_a (L/R)
      src = ahs + ((size_t)(b * 25 + (t + 1)) * 80 + 64 + arm * 8 + pos) * D;
    } else {
      src = xo + (size_t)row * D;                                 // x_other
    }
    sRow[threadIdx.x] = src;
  }
  gemm_core<false, false>(sRow, W, bias, nullptr, dst, colBase);
}

// ---------------------------------------------------------------------------
// RoPE on q, ks, ki (both arms). One thread per even/odd pair.
// ---------------------------------------------------------------------------
__global__ __launch_bounds__(256)
void rope_kernel(float* __restrict__ ws)
{
  const int id   = blockIdx.x * 256 + threadIdx.x;   // 0 .. 98303
  const int pair = id & 255;
  const int row  = (id >> 8) & 63;
  const int rest = id >> 14;                          // 0..5
  const int arm  = rest / 3;
  const int tens = rest % 3;

  const size_t toff = (tens == 0) ? OQ : (tens == 1) ? OKS : OKI;
  float* p = ws + (size_t)arm * ARM_STR + toff + (size_t)row * D;

  const int d0  = pair * 2;
  const int dd0 = d0 & 63;                 // within head
  const float posf = (float)(row & 7);
  const float j0 = (float)(dd0 & 31);
  const float j1 = (float)((dd0 + 1) & 31);
  const float a0 = posf * __expf(j0 * -0.2878231366242557f);  // ln(1e4)/32
  const float a1 = posf * __expf(j1 * -0.2878231366242557f);
  const float c0 = __cosf(a0), s0 = __sinf(a0);
  const float c1 = __cosf(a1), s1 = __sinf(a1);

  const float x0 = p[d0], x1 = p[d0 + 1];
  p[d0]     = x0 * c0 - x1 * s0;   // rotate_half: (-x_odd, x_even)
  p[d0 + 1] = x1 * c1 + x0 * s1;
}

// ---------------------------------------------------------------------------
// Attention: one block per (b,h,arm); 88-key context, softmax, A@V.
// ---------------------------------------------------------------------------
__global__ __launch_bounds__(128)
void attn_kernel(float* __restrict__ ws, const float* __restrict__ gating, int t)
{
  const int arm = blockIdx.y;
  const int b = blockIdx.x >> 3, h = blockIdx.x & 7;
  const float g = tanhf(gating[t]);
  float* base = ws + (size_t)arm * ARM_STR;
  const float *q = base + OQ, *ks = base + OKS, *vs = base + OVS,
              *kt = base + OKT, *vt = base + OVT, *ka = base + OKA,
              *va = base + OVA, *ki = base + OKI, *vi = base + OVI;
  float* aout = base + OAOUT;

  __shared__ float sQ[8 * 64], sK[KCTX * 64], sV[KCTX * 64], sS[8 * KCTX];
  const int tid = threadIdx.x;

  for (int i = tid; i < 512; i += 128) {
    const int pos = i >> 6, d = i & 63;
    sQ[i] = q[(size_t)(b * 8 + pos) * D + h * 64 + d];
  }
  for (int i = tid; i < KCTX * 64; i += 128) {
    const int c = i >> 6, d = i & 63;
    const float *kp, *vp; int row;
    if (c < 8)       { kp = ks; vp = vs; row = b * 8  + c;        }
    else if (c < 16) { kp = ka; vp = va; row = b * 8  + (c - 8);  }
    else if (c < 80) { kp = kt; vp = vt; row = b * 64 + (c - 16); }
    else             { kp = ki; vp = vi; row = b * 8  + (c - 80); }
    sK[i] = kp[(size_t)row * D + h * 64 + d];
    sV[i] = vp[(size_t)row * D + h * 64 + d];
  }
  __syncthreads();

  for (int s = tid; s < 8 * KCTX; s += 128) {
    const int qr = s / KCTX, c = s % KCTX;
    float acc = 0.0f;
#pragma unroll 8
    for (int d = 0; d < 64; ++d) acc += sQ[qr * 64 + d] * sK[c * 64 + d];
    const float scale = (c >= 16 && c < 80) ? g : 1.0f;
    sS[s] = acc * scale * 0.125f;            // 1/sqrt(64)
  }
  __syncthreads();

  if (tid < 8) {
    float mx = -1e30f;
    for (int c = 0; c < KCTX; ++c) mx = fmaxf(mx, sS[tid * KCTX + c]);
    float sum = 0.0f;
    for (int c = 0; c < KCTX; ++c) {
      const float e = __expf(sS[tid * KCTX + c] - mx);
      sS[tid * KCTX + c] = e; sum += e;
    }
    const float inv = 1.0f / sum;
    for (int c = 0; c < KCTX; ++c) sS[tid * KCTX + c] *= inv;
  }
  __syncthreads();

  for (int o = tid; o < 512; o += 128) {
    const int qr = o >> 6, d = o & 63;
    float acc = 0.0f;
    for (int c = 0; c < KCTX; ++c) acc += sS[qr * KCTX + c] * sV[c * 64 + d];
    aout[(size_t)(b * 8 + qr) * D + h * 64 + d] = acc;
  }
}

// ---------------------------------------------------------------------------
// Out-projection (W[9]) + residual:  ybuf = aout @ W9 + b9 + x_self
// ---------------------------------------------------------------------------
__global__ __launch_bounds__(128)
void outproj_kernel(const float* __restrict__ WL, const float* __restrict__ WR,
                    const float* __restrict__ bL, const float* __restrict__ bR,
                    const float* __restrict__ xL, const float* __restrict__ xR,
                    float* __restrict__ ws, int t)
{
  const int arm = blockIdx.z, colBase = blockIdx.y * 256, rt = blockIdx.x;
  const float* W    = (arm ? WR : WL) + (size_t)(t * 11 + 9) * D * D;
  const float* bias = (arm ? bR : bL) + (size_t)(t * 11 + 9) * D;
  const float* A     = ws + (size_t)arm * ARM_STR + OAOUT;
  const float* resid = (arm ? xR : xL) + (size_t)rt * 16 * D;
  float* dst = ws + (size_t)arm * ARM_STR + OYBUF + (size_t)rt * 16 * D;

  __shared__ const float* sRow[16];
  if (threadIdx.x < 16)
    sRow[threadIdx.x] = A + (size_t)(rt * 16 + threadIdx.x) * D;
  gemm_core<false, true>(sRow, W, bias, resid, dst, colBase);
}

// ---------------------------------------------------------------------------
// LayerNorm: lny = LN(ybuf) * gamma + beta
// ---------------------------------------------------------------------------
__global__ __launch_bounds__(256)
void ln_kernel(float* __restrict__ ws,
               const float* __restrict__ lngL, const float* __restrict__ lnbL,
               const float* __restrict__ lngR, const float* __restrict__ lnbR, int t)
{
  const int arm = blockIdx.y, row = blockIdx.x, tid = threadIdx.x;
  const float* gamma = (arm ? lngR : lngL) + (size_t)t * D;
  const float* beta  = (arm ? lnbR : lnbL) + (size_t)t * D;
  const float* x = ws + (size_t)arm * ARM_STR + OYBUF + (size_t)row * D;
  float* y       = ws + (size_t)arm * ARM_STR + OLNY  + (size_t)row * D;

  __shared__ float red[256];
  const float a = x[tid], bv = x[tid + 256];
  red[tid] = a + bv; __syncthreads();
  for (int o = 128; o > 0; o >>= 1) { if (tid < o) red[tid] += red[tid + o]; __syncthreads(); }
  const float mu = red[0] * (1.0f / 512.0f); __syncthreads();
  red[tid] = (a - mu) * (a - mu) + (bv - mu) * (bv - mu); __syncthreads();
  for (int o = 128; o > 0; o >>= 1) { if (tid < o) red[tid] += red[tid + o]; __syncthreads(); }
  const float rstd = rsqrtf(red[0] * (1.0f / 512.0f) + 1e-5f);
  y[tid]       = (a  - mu) * rstd * gamma[tid]       + beta[tid];
  y[tid + 256] = (bv - mu) * rstd * gamma[tid + 256] + beta[tid + 256];
}

// ---------------------------------------------------------------------------
// FFN (W[10]) + ReLU:  x_next = relu(lny @ W10 + b10)
// ---------------------------------------------------------------------------
__global__ __launch_bounds__(128)
void ffn_kernel(const float* __restrict__ WL, const float* __restrict__ WR,
                const float* __restrict__ bL, const float* __restrict__ bR,
                float* __restrict__ xnL, float* __restrict__ xnR,
                float* __restrict__ ws, int t)
{
  const int arm = blockIdx.z, colBase = blockIdx.y * 256, rt = blockIdx.x;
  const float* W    = (arm ? WR : WL) + (size_t)(t * 11 + 10) * D * D;
  const float* bias = (arm ? bR : bL) + (size_t)(t * 11 + 10) * D;
  const float* A = ws + (size_t)arm * ARM_STR + OLNY;
  float* dst = (arm ? xnR : xnL) + (size_t)rt * 16 * D;

  __shared__ const float* sRow[16];
  if (threadIdx.x < 16)
    sRow[threadIdx.x] = A + (size_t)(rt * 16 + threadIdx.x) * D;
  gemm_core<true, false>(sRow, W, bias, nullptr, dst, colBase);
}

// ---------------------------------------------------------------------------
// Final LN + 512->7 heads for both arms, concat to (B, 8, 14).
// ---------------------------------------------------------------------------
__global__ __launch_bounds__(256)
void final_kernel(const float* __restrict__ xL, const float* __restrict__ xR,
                  const float* __restrict__ fgL, const float* __restrict__ fbL,
                  const float* __restrict__ fgR, const float* __restrict__ fbR,
                  const float* __restrict__ fwL, const float* __restrict__ fcbL,
                  const float* __restrict__ fwR, const float* __restrict__ fcbR,
                  float* __restrict__ out)
{
  const int row = blockIdx.x, tid = threadIdx.x;
  __shared__ float red[256];
  __shared__ float nrm[512];

  for (int arm = 0; arm < 2; ++arm) {
    const float* x   = (arm ? xR : xL) + (size_t)row * D;
    const float* fg  = arm ? fgR : fgL;
    const float* fb  = arm ? fbR : fbL;
    const float* fw  = arm ? fwR : fwL;
    const float* fcb = arm ? fcbR : fcbL;

    const float a = x[tid], bv = x[tid + 256];
    red[tid] = a + bv; __syncthreads();
    for (int o = 128; o > 0; o >>= 1) { if (tid < o) red[tid] += red[tid + o]; __syncthreads(); }
    const float mu = red[0] * (1.0f / 512.0f); __syncthreads();
    red[tid] = (a - mu) * (a - mu) + (bv - mu) * (bv - mu); __syncthreads();
    for (int o = 128; o > 0; o >>= 1) { if (tid < o) red[tid] += red[tid + o]; __syncthreads(); }
    const float rstd = rsqrtf(red[0] * (1.0f / 512.0f) + 1e-5f); __syncthreads();
    nrm[tid]       = (a  - mu) * rstd * fg[tid]       + fb[tid];
    nrm[tid + 256] = (bv - mu) * rstd * fg[tid + 256] + fb[tid + 256];
    __syncthreads();

    for (int o = 0; o < 7; ++o) {
      red[tid] = nrm[tid] * fw[tid * 7 + o] + nrm[tid + 256] * fw[(tid + 256) * 7 + o];
      __syncthreads();
      for (int s = 128; s > 0; s >>= 1) { if (tid < s) red[tid] += red[tid + s]; __syncthreads(); }
      if (tid == 0) out[(size_t)row * 14 + arm * 7 + o] = red[0] + fcb[o];
      __syncthreads();
    }
  }
}

// ---------------------------------------------------------------------------
extern "C" void kernel_launch(void* const* d_in, const int* in_sizes, int n_in,
                              void* d_out, int out_size, void* d_ws, size_t ws_size,
                              hipStream_t stream)
{
  (void)in_sizes; (void)n_in; (void)out_size; (void)ws_size;

  const float* ahs    = (const float*)d_in[0];
  const float* W_L    = (const float*)d_in[1];
  const float* b_L    = (const float*)d_in[2];
  const float* lng_L  = (const float*)d_in[3];
  const float* lnb_L  = (const float*)d_in[4];
  const float* W_R    = (const float*)d_in[5];
  const float* b_R    = (const float*)d_in[6];
  const float* lng_R  = (const float*)d_in[7];
  const float* lnb_R  = (const float*)d_in[8];
  const float* gating = (const float*)d_in[9];
  const float* fg_L   = (const float*)d_in[10];
  const float* fb_L   = (const float*)d_in[11];
  const float* fg_R   = (const float*)d_in[12];
  const float* fb_R   = (const float*)d_in[13];
  const float* fw_L   = (const float*)d_in[14];
  const float* fcb_L  = (const float*)d_in[15];
  const float* fw_R   = (const float*)d_in[16];
  const float* fcb_R  = (const float*)d_in[17];
  float* out = (float*)d_out;
  float* ws  = (float*)d_ws;

  float* xbuf = ws + 2 * ARM_STR;
  float* xc[2] = { xbuf,            xbuf + XSZ };       // current xL, xR
  float* xn[2] = { xbuf + 2 * XSZ,  xbuf + 3 * XSZ };   // next xL, xR

  hipMemsetAsync(xc[0], 0, 2 * XSZ * sizeof(float), stream);  // x0 = zeros

  for (int t = 0; t < NBLK; ++t) {
    proj9_kernel  <<<dim3(92, 2, 2), 128, 0, stream>>>(ahs, W_L, W_R, b_L, b_R, xc[0], xc[1], ws, t);
    rope_kernel   <<<384, 256, 0, stream>>>(ws);
    attn_kernel   <<<dim3(64, 2), 128, 0, stream>>>(ws, gating, t);
    outproj_kernel<<<dim3(4, 2, 2), 128, 0, stream>>>(W_L, W_R, b_L, b_R, xc[0], xc[1], ws, t);
    ln_kernel     <<<dim3(64, 2), 256, 0, stream>>>(ws, lng_L, lnb_L, lng_R, lnb_R, t);
    ffn_kernel    <<<dim3(4, 2, 2), 128, 0, stream>>>(W_L, W_R, b_L, b_R, xn[0], xn[1], ws, t);
    float* t0 = xc[0]; float* t1 = xc[1];
    xc[0] = xn[0]; xc[1] = xn[1]; xn[0] = t0; xn[1] = t1;
  }

  final_kernel<<<64, 256, 0, stream>>>(xc[0], xc[1], fg_L, fb_L, fg_R, fb_R,
                                       fw_L, fcb_L, fw_R, fcb_R, out);
}
